// Attention4D_64819646431287
// MI455X (gfx1250) — matrix operand order repaired and checked by reading, not compiled, not run on hardware
//
#include <hip/hip_runtime.h>

typedef __attribute__((ext_vector_type(16))) __bf16 v16bf;
typedef __attribute__((ext_vector_type(8)))  __bf16 v8bf;
typedef __attribute__((ext_vector_type(8)))  float  v8f;

constexpr int B_   = 256;
constexpr int C_   = 384;
constexpr int H_   = 14;
constexpr int N_   = 196;    // 14*14
constexpr int P_   = 224;    // N padded to 14 tiles of 16 / 7 K-steps of 32
constexpr int NH_  = 8;
constexpr int KD_  = 32;
constexpr int D_   = 128;
constexpr int DH_  = 1024;
constexpr int MQKV_ = 1536;  // 256(q)+256(k)+1024(v)
constexpr float SCALE_ = 0.17677669529663689f; // 32^-0.5
constexpr float EPS_   = 1e-5f;

// ---- WMMA bf16 16x16x32 fragment loaders (ISA 7.12.2 layouts) ----
// A (16x32): lane m = lane%16; element i -> k = (i/8)*16 + (lane/16)*8 + i%8
__device__ __forceinline__ v16bf load_frag_a(const __bf16* base, int hi) {
  v8bf lo = *(const v8bf*)(base + hi * 8);
  v8bf up = *(const v8bf*)(base + 16 + hi * 8);
  return __builtin_shufflevector(lo, up, 0,1,2,3,4,5,6,7,8,9,10,11,12,13,14,15);
}
// B (32x16): lane n = lane%16; element i -> k = (i/8)*16 + (lane/16)*8 + i%8
__device__ __forceinline__ v16bf load_frag_b(const __bf16* base, int hi) {
  v8bf lo = *(const v8bf*)(base + hi * 8);
  v8bf up = *(const v8bf*)(base + 16 + hi * 8);
  return __builtin_shufflevector(lo, up, 0,1,2,3,4,5,6,7,8,9,10,11,12,13,14,15);
}
// C/D (16x16 f32): element r -> row = r + 8*(lane/16), col = lane%16

// ============================ BN folding ============================
__global__ void fold_qkv(const float* wq, const float* bq, const float* qg, const float* qbe, const float* qm, const float* qv,
                         const float* wk, const float* bk, const float* kg, const float* kbe, const float* km, const float* kv,
                         const float* wv, const float* bv, const float* vg, const float* vbe, const float* vm, const float* vv,
                         __bf16* W, float* bias) {
  int idx = blockIdx.x * blockDim.x + threadIdx.x;
  if (idx >= MQKV_ * C_) return;
  int o = idx / C_, c = idx % C_;
  const float *w, *b, *g, *be, *m, *v; int oo;
  if (o < 256)      { w = wq; b = bq; g = qg; be = qbe; m = qm; v = qv; oo = o; }
  else if (o < 512) { w = wk; b = bk; g = kg; be = kbe; m = km; v = kv; oo = o - 256; }
  else              { w = wv; b = bv; g = vg; be = vbe; m = vm; v = vv; oo = o - 512; }
  float s = g[oo] * rsqrtf(v[oo] + EPS_);
  W[(size_t)o * C_ + c] = (__bf16)(w[(size_t)oo * C_ + c] * s);
  if (c == 0) bias[o] = (b[oo] - m[oo]) * s + be[oo];
}

__global__ void fold_p(const float* wp, const float* bp, const float* pg, const float* pbe, const float* pm, const float* pv,
                       __bf16* W, float* bias) {
  int idx = blockIdx.x * blockDim.x + threadIdx.x;
  if (idx >= C_ * DH_) return;
  int o = idx / DH_, c = idx % DH_;
  float s = pg[o] * rsqrtf(pv[o] + EPS_);
  W[idx] = (__bf16)(wp[idx] * s);
  if (c == 0) bias[o] = (bp[o] - pm[o]) * s + pbe[o];
}

__global__ void fold_vl(const float* wvl, const float* bvl, const float* g, const float* be, const float* m, const float* v,
                        float* W, float* bias) {
  int idx = blockIdx.x * blockDim.x + threadIdx.x;
  if (idx >= DH_ * 9) return;
  int ch = idx / 9;
  float s = g[ch] * rsqrtf(v[ch] + EPS_);
  W[idx] = wvl[idx] * s;
  if ((idx % 9) == 0) bias[ch] = (bvl[ch] - m[ch]) * s + be[ch];
}

// x [b][c][n] f32  ->  Xbf [b][P][C] bf16 (k-contiguous for B fragments)
__global__ void conv_x(const float* __restrict__ x, __bf16* __restrict__ X) {
  size_t idx = (size_t)blockIdx.x * blockDim.x + threadIdx.x;
  if (idx >= (size_t)B_ * N_ * C_) return;
  int c = (int)(idx % C_);
  int n = (int)((idx / C_) % N_);
  int b = (int)(idx / ((size_t)C_ * N_));
  X[((size_t)b * P_ + n) * C_ + c] = (__bf16)x[((size_t)b * C_ + c) * N_ + n];
}

// ---- zero-fill padded regions (never touched by compute kernels) ----
__global__ void zpad_rows(__bf16* buf, int nrow, int rowlen) { // rows [196, P)
  size_t idx = (size_t)blockIdx.x * blockDim.x + threadIdx.x;
  size_t total = (size_t)nrow * 28 * rowlen;
  if (idx >= total) return;
  int c = (int)(idx % rowlen);
  int j = (int)((idx / rowlen) % 28);
  size_t r = idx / ((size_t)rowlen * 28);
  buf[(r * P_ + N_ + j) * rowlen + c] = (__bf16)0.f;
}
__global__ void zpad_cols(__bf16* buf, size_t nrow) {          // cols [196, P)
  size_t idx = (size_t)blockIdx.x * blockDim.x + threadIdx.x;
  if (idx >= nrow * 28) return;
  int j = (int)(idx % 28);
  size_t r = idx / 28;
  buf[r * P_ + N_ + j] = (__bf16)0.f;
}

// ============================ QKV GEMM ============================
// Per batch: Out[1536,224] = Wqkv[1536,384] @ Xbf^T + bias.
// Wave computes a 4Mx2N block of 16x16 tiles: 12 b128 loads -> 8 WMMAs / K-step.
// D scattered to transposed consumer layouts Q[n][kd], K[m][kd], V[d][m].
__global__ __launch_bounds__(256)
void qkv_gemm(const __bf16* __restrict__ W, const float* __restrict__ bias,
              const __bf16* __restrict__ X,
              __bf16* __restrict__ Qb, __bf16* __restrict__ Kb, __bf16* __restrict__ Vb) {
  int b    = blockIdx.x;
  int wave = threadIdx.x >> 5;
  int lane = threadIdx.x & 31;
  int hi = lane >> 4, lm = lane & 15;
  int tile = blockIdx.y * 8 + wave;      // 24 mt4 * 7 nt2 = 168 tiles, grid.y = 21
  int mt4 = tile / 7, nt2 = tile % 7;
  int m0 = mt4 * 64, n0 = nt2 * 32;
  const __bf16* Xb = X + (size_t)b * P_ * C_;
  const __bf16* arow0 = W + (size_t)(m0 + lm) * C_;
  const __bf16* brow0 = Xb + (size_t)(n0 + lm) * C_;
  const __bf16* brow1 = brow0 + (size_t)16 * C_;
  v8f acc[4][2] = {};
  for (int kt = 0; kt < 12; ++kt) {
    v16bf bf0 = load_frag_b(brow0 + kt * 32, hi);
    v16bf bf1 = load_frag_b(brow1 + kt * 32, hi);
#pragma unroll
    for (int j = 0; j < 4; ++j) {
      v16bf af = load_frag_a(arow0 + (size_t)j * 16 * C_ + kt * 32, hi);
      acc[j][0] = __builtin_amdgcn_wmma_f32_16x16x32_bf16(false, af, false, bf0, (short)0, acc[j][0], false, false);
      acc[j][1] = __builtin_amdgcn_wmma_f32_16x16x32_bf16(false, af, false, bf1, (short)0, acc[j][1], false, false);
    }
  }
#pragma unroll
  for (int t = 0; t < 2; ++t) {
    int n = n0 + t * 16 + lm;
    if (n >= N_) continue;
#pragma unroll
    for (int j = 0; j < 4; ++j) {
      int obase = m0 + j * 16 + hi * 8;  // segment uniform per (j, hi)
#pragma unroll
      for (int r = 0; r < 8; ++r) {
        int o = obase + r;
        __bf16 val = (__bf16)(acc[j][t][r] + bias[o]);
        if (o < 256) {
          int h = o >> 5, kd = o & 31;
          Qb[(((size_t)b * NH_ + h) * P_ + n) * KD_ + kd] = val;
        } else if (o < 512) {
          int o2 = o - 256, h = o2 >> 5, kd = o2 & 31;
          Kb[(((size_t)b * NH_ + h) * P_ + n) * KD_ + kd] = val;
        } else {
          int o2 = o - 512, h = o2 >> 7, d = o2 & 127;
          Vb[(((size_t)b * NH_ + h) * D_ + d) * P_ + n] = val;
        }
      }
    }
  }
}

// ===================== L1 normalization (in place) =====================
__global__ void l1norm(__bf16* __restrict__ buf, float post) {
  int idx = blockIdx.x * blockDim.x + threadIdx.x;
  if (idx >= B_ * NH_ * N_) return;
  int n = idx % N_;
  int bh = idx / N_;
  __bf16* p = buf + ((size_t)bh * P_ + n) * KD_;
  v8bf v[4];
  float s = 0.f;
#pragma unroll
  for (int j = 0; j < 4; ++j) {
    v[j] = ((const v8bf*)p)[j];
#pragma unroll
    for (int e = 0; e < 8; ++e) s += fabsf((float)v[j][e]);
  }
  float inv = post / fmaxf(s, 1e-12f);
#pragma unroll
  for (int j = 0; j < 4; ++j) {
    v8bf o;
#pragma unroll
    for (int e = 0; e < 8; ++e) o[e] = (__bf16)((float)v[j][e] * inv);
    ((v8bf*)p)[j] = o;
  }
}

// ============================ Attention ============================
// Per (b,h): attn = qn^T kn (kd=32: one WMMA/tile, q fragment reused over all
// 14 column tiles; SCALE folded into q); stage attn rows bf16 in wave-private
// LDS; hoist the 7 attn A-fragments and reuse across all 8 d-tiles.
__global__ __launch_bounds__(256)
void attn_kernel(const __bf16* __restrict__ Qb, const __bf16* __restrict__ Kb,
                 const __bf16* __restrict__ Vb, __bf16* __restrict__ Yattn) {
  int b = blockIdx.x;
  int h = blockIdx.y;
  const __bf16* Q = Qb + ((size_t)b * NH_ + h) * P_ * KD_;   // [224][32]
  const __bf16* K = Kb + ((size_t)b * NH_ + h) * P_ * KD_;   // [224][32]
  const __bf16* V = Vb + ((size_t)b * NH_ + h) * D_ * P_;    // [128][224]

  __shared__ __align__(16) __bf16 sAttn[8][16][P_];  // wave-private slices

  int wave = threadIdx.x >> 5, lane = threadIdx.x & 31;
  int hi = lane >> 4, lm = lane & 15;

  for (int nt = wave; nt < 13; nt += 8) {
    // --- attn row tile [16 x 224] ---
    v16bf aq = load_frag_a(Q + (size_t)(nt * 16 + lm) * KD_, hi);
    for (int mt = 0; mt < 14; ++mt) {
      v16bf bk = load_frag_b(K + (size_t)(mt * 16 + lm) * KD_, hi);
      v8f acc = {};
      acc = __builtin_amdgcn_wmma_f32_16x16x32_bf16(false, aq, false, bk, (short)0, acc, false, false);
#pragma unroll
      for (int r = 0; r < 8; ++r)
        sAttn[wave][hi * 8 + r][mt * 16 + lm] = (__bf16)acc[r];
    }
    // --- load the 7 attn A-fragments once ---
    v16bf aa[7];
#pragma unroll
    for (int kt = 0; kt < 7; ++kt)
      aa[kt] = load_frag_a(&sAttn[wave][lm][kt * 32], hi);
    // --- out tile [16 x 128] = attn[16 x 224] @ v[224 x 128] ---
    for (int dt = 0; dt < 8; ++dt) {
      int d = dt * 16 + lm;
      const __bf16* vrow = V + (size_t)d * P_;
      v8f acc = {};
#pragma unroll
      for (int kt = 0; kt < 7; ++kt) {
        v16bf bv = load_frag_b(vrow + kt * 32, hi);
        acc = __builtin_amdgcn_wmma_f32_16x16x32_bf16(false, aa[kt], false, bv, (short)0, acc, false, false);
      }
#pragma unroll
      for (int r = 0; r < 8; ++r) {
        int n = nt * 16 + hi * 8 + r;
        if (n < N_)
          Yattn[((size_t)b * N_ + n) * DH_ + h * D_ + d] = (__bf16)acc[r];
      }
    }
  }
}

// ==================== dwconv3x3 + add + ReLU ====================
__global__ void fuse_kernel(const __bf16* __restrict__ Vb, const __bf16* __restrict__ Yattn,
                            const float* __restrict__ Wvl, const float* __restrict__ Bvl,
                            __bf16* __restrict__ Y) {
  size_t idx = (size_t)blockIdx.x * blockDim.x + threadIdx.x;
  if (idx >= (size_t)B_ * N_ * DH_) return;
  int ch = (int)(idx % DH_);
  int n  = (int)((idx / DH_) % N_);
  int b  = (int)(idx / ((size_t)DH_ * N_));
  int h = ch >> 7, d = ch & 127;
  int yy = n / H_, xx = n % H_;
  const __bf16* Vm = Vb + (((size_t)b * NH_ + h) * D_ + d) * P_;
  float acc = Bvl[ch];
#pragma unroll
  for (int dy = -1; dy <= 1; ++dy)
#pragma unroll
    for (int dx = -1; dx <= 1; ++dx) {
      int y2 = yy + dy, x2 = xx + dx;
      if (y2 >= 0 && y2 < H_ && x2 >= 0 && x2 < H_)
        acc += (float)Vm[y2 * H_ + x2] * Wvl[ch * 9 + (dy + 1) * 3 + (dx + 1)];
    }
  float o = (float)Yattn[idx] + acc;
  Y[((size_t)b * P_ + n) * DH_ + ch] = (__bf16)fmaxf(o, 0.f);
}

// ============================ Projection GEMM ============================
// Per batch: out[384,196] = Wp[384,1024] @ Y^T + bias, f32 out. 4Mx2N blocking.
__global__ __launch_bounds__(256)
void proj_gemm(const __bf16* __restrict__ W, const float* __restrict__ bias,
               const __bf16* __restrict__ Y, float* __restrict__ out) {
  int b    = blockIdx.x;
  int wave = threadIdx.x >> 5;
  int lane = threadIdx.x & 31;
  int hi = lane >> 4, lm = lane & 15;
  int tile = blockIdx.y * 8 + wave;    // 6 mt4 * 7 nt2 = 42 tiles, grid.y = 6
  if (tile >= 42) return;
  int mt4 = tile / 7, nt2 = tile % 7;
  int m0 = mt4 * 64, n0 = nt2 * 32;
  const __bf16* Yb = Y + (size_t)b * P_ * DH_;
  const __bf16* arow0 = W + (size_t)(m0 + lm) * DH_;
  const __bf16* brow0 = Yb + (size_t)(n0 + lm) * DH_;
  const __bf16* brow1 = brow0 + (size_t)16 * DH_;
  v8f acc[4][2] = {};
  for (int kt = 0; kt < 32; ++kt) {
    v16bf bf0 = load_frag_b(brow0 + kt * 32, hi);
    v16bf bf1 = load_frag_b(brow1 + kt * 32, hi);
#pragma unroll
    for (int j = 0; j < 4; ++j) {
      v16bf af = load_frag_a(arow0 + (size_t)j * 16 * DH_ + kt * 32, hi);
      acc[j][0] = __builtin_amdgcn_wmma_f32_16x16x32_bf16(false, af, false, bf0, (short)0, acc[j][0], false, false);
      acc[j][1] = __builtin_amdgcn_wmma_f32_16x16x32_bf16(false, af, false, bf1, (short)0, acc[j][1], false, false);
    }
  }
#pragma unroll
  for (int t = 0; t < 2; ++t) {
    int n = n0 + t * 16 + lm;
    if (n >= N_) continue;
#pragma unroll
    for (int j = 0; j < 4; ++j) {
#pragma unroll
      for (int r = 0; r < 8; ++r) {
        int o = m0 + j * 16 + hi * 8 + r;
        out[((size_t)b * C_ + o) * N_ + n] = acc[j][t][r] + bias[o];
      }
    }
  }
}

// ============================ Launch ============================
extern "C" void kernel_launch(void* const* d_in, const int* in_sizes, int n_in,
                              void* d_out, int out_size, void* d_ws, size_t ws_size,
                              hipStream_t stream) {
  const float* x    = (const float*)d_in[0];
  const float* wq   = (const float*)d_in[1];
  const float* bq   = (const float*)d_in[2];
  const float* qg   = (const float*)d_in[3];
  const float* qbe  = (const float*)d_in[4];
  const float* qm   = (const float*)d_in[5];
  const float* qv   = (const float*)d_in[6];
  const float* wk   = (const float*)d_in[7];
  const float* bk   = (const float*)d_in[8];
  const float* kg   = (const float*)d_in[9];
  const float* kbe  = (const float*)d_in[10];
  const float* km   = (const float*)d_in[11];
  const float* kv   = (const float*)d_in[12];
  const float* wv   = (const float*)d_in[13];
  const float* bv   = (const float*)d_in[14];
  const float* vg   = (const float*)d_in[15];
  const float* vbe  = (const float*)d_in[16];
  const float* vm   = (const float*)d_in[17];
  const float* vv   = (const float*)d_in[18];
  const float* wvl  = (const float*)d_in[19];
  const float* bvl  = (const float*)d_in[20];
  const float* vlg  = (const float*)d_in[21];
  const float* vlbe = (const float*)d_in[22];
  const float* vlm  = (const float*)d_in[23];
  const float* vlv  = (const float*)d_in[24];
  const float* wp   = (const float*)d_in[25];
  const float* bp   = (const float*)d_in[26];
  const float* pg   = (const float*)d_in[27];
  const float* pbe  = (const float*)d_in[28];
  const float* pm   = (const float*)d_in[29];
  const float* pv   = (const float*)d_in[30];

  char* ws = (char*)d_ws;
  size_t off = 0;
  auto alloc = [&](size_t bytes) -> void* {
    void* p = ws + off;
    off = (off + bytes + 255) & ~(size_t)255;
    return p;
  };
  __bf16* Wqkv  = (__bf16*)alloc((size_t)MQKV_ * C_ * 2);
  float*  Bqkv  = (float*) alloc((size_t)MQKV_ * 4);
  __bf16* Wp    = (__bf16*)alloc((size_t)C_ * DH_ * 2);
  float*  Bp    = (float*) alloc((size_t)C_ * 4);
  float*  Wvl   = (float*) alloc((size_t)DH_ * 9 * 4);
  float*  Bvl   = (float*) alloc((size_t)DH_ * 4);
  __bf16* Xbf   = (__bf16*)alloc((size_t)B_ * P_ * C_ * 2);
  __bf16* Qb    = (__bf16*)alloc((size_t)B_ * NH_ * P_ * KD_ * 2);
  __bf16* Kb    = (__bf16*)alloc((size_t)B_ * NH_ * P_ * KD_ * 2);
  __bf16* Vb    = (__bf16*)alloc((size_t)B_ * NH_ * D_ * P_ * 2);
  __bf16* Yattn = (__bf16*)alloc((size_t)B_ * N_ * DH_ * 2);
  __bf16* Y     = (__bf16*)alloc((size_t)B_ * P_ * DH_ * 2);
  (void)ws_size; (void)in_sizes; (void)n_in; (void)out_size;

  fold_qkv<<<(MQKV_ * C_ + 255) / 256, 256, 0, stream>>>(
      wq, bq, qg, qbe, qm, qv, wk, bk, kg, kbe, km, kv, wv, bv, vg, vbe, vm, vv, Wqkv, Bqkv);
  fold_p<<<(C_ * DH_ + 255) / 256, 256, 0, stream>>>(wp, bp, pg, pbe, pm, pv, Wp, Bp);
  fold_vl<<<(DH_ * 9 + 255) / 256, 256, 0, stream>>>(wvl, bvl, vlg, vlbe, vlm, vlv, Wvl, Bvl);
  conv_x<<<(int)(((size_t)B_ * N_ * C_ + 255) / 256), 256, 0, stream>>>(x, Xbf);

  zpad_rows<<<(int)(((size_t)B_ * 28 * C_ + 255) / 256), 256, 0, stream>>>(Xbf, B_, C_);
  zpad_rows<<<(int)(((size_t)B_ * NH_ * 28 * KD_ + 255) / 256), 256, 0, stream>>>(Qb, B_ * NH_, KD_);
  zpad_rows<<<(int)(((size_t)B_ * NH_ * 28 * KD_ + 255) / 256), 256, 0, stream>>>(Kb, B_ * NH_, KD_);
  zpad_cols<<<(int)(((size_t)B_ * NH_ * D_ * 28 + 255) / 256), 256, 0, stream>>>(Vb, (size_t)B_ * NH_ * D_);
  zpad_rows<<<(int)(((size_t)B_ * 28 * DH_ + 255) / 256), 256, 0, stream>>>(Y, B_, DH_);

  qkv_gemm<<<dim3(B_, 21), 256, 0, stream>>>(Wqkv, Bqkv, Xbf, Qb, Kb, Vb);

  l1norm<<<(B_ * NH_ * N_ + 255) / 256, 256, 0, stream>>>(Qb, SCALE_);
  l1norm<<<(B_ * NH_ * N_ + 255) / 256, 256, 0, stream>>>(Kb, 1.0f);

  attn_kernel<<<dim3(B_, NH_), 256, 0, stream>>>(Qb, Kb, Vb, Yattn);

  fuse_kernel<<<(int)(((size_t)B_ * N_ * DH_ + 255) / 256), 256, 0, stream>>>(Vb, Yattn, Wvl, Bvl, Y);

  proj_gemm<<<dim3(B_, 6), 256, 0, stream>>>(Wp, Bp, Y, (float*)d_out);
}
